// Net_29557964931415
// MI455X (gfx1250) — matrix operand order repaired and checked by reading, not compiled, not run on hardware
//
#include <hip/hip_runtime.h>
#include <hip/hip_bf16.h>

typedef __bf16 bf16;
typedef __attribute__((ext_vector_type(16))) __bf16 v16bf;
typedef __attribute__((ext_vector_type(8)))  __bf16 v8bf;
typedef __attribute__((ext_vector_type(8)))  float   v8f;
typedef __attribute__((ext_vector_type(4)))  unsigned int v4u;
typedef __attribute__((ext_vector_type(8)))  int v8i;
typedef __attribute__((ext_vector_type(4)))  int v4i;

#define NINP   300
#define NHID   64
#define SLEN   512
#define BSZ    256

// Workspace layout (byte offsets). ws base from hipMalloc is 256B aligned.
#define WS_WCOMB 0         // 10 ksteps * 16 ntiles * 32 lanes * 16 bf16 = 163840 B
#define WS_WGLT  163840    //  2 ksteps * 16 ntiles * 32 lanes * 16 bf16 =  32768 B
#define WS_WDEC  196608    //  2 ksteps *  1 ntile  * 32 lanes * 16 bf16 =   2048 B
#define WS_BIAS  198656    //  256 f32 (fused/permuted b_pt0|b_pt1)      =   1024 B
#define WS_I2H   199680    //  S*B*256 f32 = 134217728 B

#define OUT_HT 393216      // 512*256*3
#define OUT_CT 409600

__device__ __forceinline__ v8f wmma_bf16(v16bf a, v16bf b, v8f c) {
  // D = A(16x32 bf16) * B(32x16 bf16) + C(16x16 f32)
  return __builtin_amdgcn_wmma_f32_16x16x32_bf16(false, a, false, b, (short)0, c,
                                                 false, false);
}
// Fast activations: v_exp_f32 + v_rcp_f32, no library calls, saturate correctly
// at +/-inf (exp(-inf)=0, rcp(inf)=0).
__device__ __forceinline__ float fast_sig(float x) {
  return __builtin_amdgcn_rcpf(1.0f + __expf(-x));
}
__device__ __forceinline__ float fast_tanh(float x) {
  return 1.0f - 2.0f * __builtin_amdgcn_rcpf(1.0f + __expf(2.0f * x));
}

// Load an A-fragment (16x32 bf16) for this lane from a row-major bf16 matrix.
// p points at rowBase + kstep*32 + (lane>=16 ? 8 : 0).
// Elements 0..7 = K+0..7, elements 8..15 = K+16..23 (CDNA5 16-bit A layout).
__device__ __forceinline__ v16bf load_a_frag(const bf16* p) {
  v8bf lo = *(const v8bf*)p;          // 16B aligned
  v8bf hh = *(const v8bf*)(p + 16);   // +32 bytes
  v16bf a;
#pragma unroll
  for (int e = 0; e < 8; ++e) { a[e] = lo[e]; a[8 + e] = hh[e]; }
  return a;
}

// ---------------------------------------------------------------------------
// Kernel 1: pack weights into per-lane WMMA B-fragment order (bf16).
// B-frag layout per CDNA5 16-bit B: lane L<16 -> col N=L, K=0..7,16..23 (elems 0..15);
// lane L+16 -> col N=L, K=8..15,24..31. Fragment linear index = ((k*NT+nt)*32+lane)*16+e.
// W_comb columns pre-permuted into i2h gate order; x1-average folded into rows
// (W1exp[K] = 0.5*W_pt1[K/2]); rows K>=300 zero (K padded 300->320).
// ---------------------------------------------------------------------------
__global__ __launch_bounds__(256) void pack_weights(
    const float* __restrict__ W_pt0, const float* __restrict__ b_pt0,
    const float* __restrict__ W_pt1, const float* __restrict__ b_pt1,
    const float* __restrict__ W_glt, const float* __restrict__ W_dec,
    char* __restrict__ ws) {
  int tid = blockIdx.x * blockDim.x + threadIdx.x;
  bf16*  wcomb = (bf16*)(ws + WS_WCOMB);
  bf16*  wglt  = (bf16*)(ws + WS_WGLT);
  bf16*  wdec  = (bf16*)(ws + WS_WDEC);
  float* bias  = (float*)(ws + WS_BIAS);

  if (tid < 81920) {                       // W_comb fragments: 10*16*32*16
    int e = tid & 15, lane = (tid >> 4) & 31, nt = (tid >> 9) & 15, k = tid >> 13;
    int K = k * 32 + ((lane < 16) ? 0 : 8) + ((e < 8) ? e : e + 8);
    int m = nt * 16 + (lane & 15);
    int gate = m >> 6, w = m & 63;
    float v = 0.0f;
    if (K < NINP) {
      if (w < 32) v = W_pt0[K * 128 + gate * 32 + w];
      else        v = 0.5f * W_pt1[(K >> 1) * 128 + gate * 32 + (w - 32)];
    }
    wcomb[tid] = (bf16)v;
  } else if (tid < 81920 + 16384) {        // W_glt fragments: 2*16*32*16
    int id = tid - 81920;
    int e = id & 15, lane = (id >> 4) & 31, nt = (id >> 9) & 15, k = id >> 13;
    int i = k * 32 + ((lane < 16) ? 0 : 8) + ((e < 8) ? e : e + 8);
    int col = nt * 16 + (lane & 15);       // h2h output is already gate*64+n order
    wglt[id] = (bf16)W_glt[i * 256 + col];
  } else if (tid < 81920 + 16384 + 1024) { // W_dec fragments (N padded 3->16)
    int id = tid - (81920 + 16384);
    int e = id & 15, lane = (id >> 4) & 31, k = id >> 9;
    int K = k * 32 + ((lane < 16) ? 0 : 8) + ((e < 8) ? e : e + 8);
    int n = lane & 15;
    wdec[id] = (bf16)((n < 3) ? W_dec[K * 3 + n] : 0.0f);
  } else if (tid < 81920 + 16384 + 1024 + 256) { // fused bias (gate order)
    int m = tid - (81920 + 16384 + 1024);
    int gate = m >> 6, w = m & 63;
    bias[m] = (w < 32) ? b_pt0[gate * 32 + w] : b_pt1[gate * 32 + (w - 32)];
  }
}

// ---------------------------------------------------------------------------
// Kernel 2: i2h GEMM. (S*B x 300) * (300 x 256) + bias -> f32 i2h, gate order.
// Block = 64 rows x 256 cols, 4 waves (one 16-row strip each).
// The 64x300 f32 x-tile is DMA'd into LDS by the Tensor Data Mover (wave 0,
// TENSORcnt); tensor_dim0=300 < tile_dim0=320 so TDM zero-fills the K padding.
// Waves build A-fragments from LDS (aligned b128) with f32->bf16 convert; B
// fragments are 32B aligned loads from packed weights (L2-resident).
// 10 ksteps x 16 ntiles of v_wmma_f32_16x16x32_bf16 per wave.
// ---------------------------------------------------------------------------
__global__ __launch_bounds__(128) void i2h_gemm(const float* __restrict__ x,
                                                char* __restrict__ ws) {
  const bf16*  wcomb = (const bf16*)(ws + WS_WCOMB);
  const float* bias  = (const float*)(ws + WS_BIAS);
  float*       i2h   = (float*)(ws + WS_I2H);

  __shared__ float xsh[64 * 320];          // 80 KB block tile (K padded to 320)

  int w    = threadIdx.x >> 5;
  int lane = threadIdx.x & 31;
  int ln   = lane & 15, hi = lane >> 4;
  int r0   = blockIdx.x * 64 + w * 16;

  if (threadIdx.x < 32) {                  // wave 0 issues the tensor DMA
    unsigned lds_base = (unsigned)(uintptr_t)(void*)&xsh[0]; // low 32b = LDS addr
    unsigned long long ga =
        (unsigned long long)(uintptr_t)(x + (size_t)blockIdx.x * 64 * NINP);
    v4u g0;
    g0[0] = 1u;                                        // count=1, user descriptor
    g0[1] = lds_base;                                  // lds_addr
    g0[2] = (unsigned)(ga & 0xFFFFFFFFu);              // global_addr[31:0]
    g0[3] = (unsigned)((ga >> 32) & 0x1FFFFFFu) | (2u << 30); // addr[56:32]|type=2
    v8i g1;
    g1[0] = (int)(2u << 16);                 // data_size=4B, workgroup_mask=0
    g1[1] = (int)((300u & 0xFFFFu) << 16);   // tensor_dim0=300 (bits 79:48)
    g1[2] = 0;                               // dim0 hi=0, tensor_dim1 lo16=0
    g1[3] = (int)(2u | (320u << 16));        // tensor_dim1 hi (131072>>16), tile_dim0=320
    g1[4] = 64;                              // tile_dim1=64, tile_dim2=0
    g1[5] = 300;                             // tensor_dim0_stride=300
    g1[6] = 0;
    g1[7] = 0;
    v4i g2 = {0, 0, 0, 0};
    v4i g3 = {0, 0, 0, 0};
    v8i g4x = {0, 0, 0, 0, 0, 0, 0, 0};      // extra group (clang-23 6-arg form)
    __builtin_amdgcn_tensor_load_to_lds(g0, g1, g2, g3, g4x, 0);
    __builtin_amdgcn_s_wait_tensorcnt(0);
  }
  __syncthreads();                          // tile visible to all waves

  v8f acc[16];
#pragma unroll
  for (int nt = 0; nt < 16; ++nt) {
    float bv = bias[nt * 16 + ln];
#pragma unroll
    for (int j = 0; j < 8; ++j) acc[nt][j] = bv;
  }

  const float* xrow = xsh + (size_t)(w * 16 + ln) * 320;
  for (int k = 0; k < 10; ++k) {
    int kb = k * 32 + hi * 8;               // base K for this lane group
    const float4* p = (const float4*)(xrow + kb);
    float4 A0 = p[0], A1 = p[1], A2 = p[4], A3 = p[5];   // K+0..7, K+16..23
    float xf[16] = {A0.x, A0.y, A0.z, A0.w, A1.x, A1.y, A1.z, A1.w,
                    A2.x, A2.y, A2.z, A2.w, A3.x, A3.y, A3.z, A3.w};
    v16bf a;
#pragma unroll
    for (int e = 0; e < 16; ++e) a[e] = (bf16)xf[e];

#pragma unroll
    for (int nt = 0; nt < 16; ++nt) {
      v16bf b = *(const v16bf*)(wcomb + (((size_t)(k * 16 + nt) * 32 + lane) << 4));
      acc[nt] = wmma_bf16(a, b, acc[nt]);
    }
  }

#pragma unroll
  for (int nt = 0; nt < 16; ++nt) {
    int col = nt * 16 + ln;
#pragma unroll
    for (int j = 0; j < 8; ++j) {
      int row = r0 + (hi ? 8 + j : j);      // C layout: lanes 16..31 hold M=8..15
      i2h[(size_t)row * 256 + col] = acc[nt][j];
    }
  }
}

// ---------------------------------------------------------------------------
// Kernel 3: sequential LSTM scan. 16 blocks x 16 batch rows, 4 waves/block.
// Wave w owns hidden cols [w*16, w*16+16): per step, 4 gate tiles x 2 ksteps of
// WMMA with i2h preloaded into the accumulator (C operand). h shared via a
// DOUBLE-BUFFERED 2x2KB LDS tile -> ONE barrier per step. i2h loads for t+1
// are software-pipelined over step t. Decoder fused as 2 WMMAs on wave 0.
// ---------------------------------------------------------------------------
__global__ __launch_bounds__(128) void lstm_scan(const float* __restrict__ h0,
                                                 const float* __restrict__ c0,
                                                 const float* __restrict__ b_dec,
                                                 const char* __restrict__ ws,
                                                 float* __restrict__ out) {
  const bf16*  wgltf = (const bf16*)(ws + WS_WGLT);
  const bf16*  wdecf = (const bf16*)(ws + WS_WDEC);
  const float* i2h   = (const float*)(ws + WS_I2H);
  __shared__ bf16 hsh[2][16 * NHID];       // double-buffered h tile (bf16)

  int w    = threadIdx.x >> 5;
  int lane = threadIdx.x & 31;
  int ln   = lane & 15, hi = lane >> 4;
  int b0   = blockIdx.x * 16;

  // Persistent W_glt B-fragments: gate g, kstep k -> ntile g*4+w.
  v16bf bglt[4][2];
#pragma unroll
  for (int g = 0; g < 4; ++g)
#pragma unroll
    for (int k = 0; k < 2; ++k)
      bglt[g][k] =
          *(const v16bf*)(wgltf + (((size_t)(k * 16 + (g * 4 + w)) * 32 + lane) << 4));
  v16bf bdec0 = *(const v16bf*)(wdecf + (((size_t)0 * 32 + lane) << 4));
  v16bf bdec1 = *(const v16bf*)(wdecf + (((size_t)1 * 32 + lane) << 4));
  float decb  = (ln < 3) ? b_dec[ln] : 0.0f;

  // c state lives in the C-operand register layout.
  v8f c;
#pragma unroll
  for (int j = 0; j < 8; ++j) {
    int row = hi ? 8 + j : j;
    c[j] = c0[(size_t)(b0 + row) * NHID + w * 16 + ln];
  }

  // h0 -> LDS buffer 0 (bf16)
  for (int i = threadIdx.x; i < 16 * NHID; i += 128)
    hsh[0][i] = (bf16)h0[(size_t)b0 * NHID + i];
  __syncthreads();

  const bf16* arow0 = &hsh[0][ln * NHID + (hi ? 8 : 0)];
  const bf16* arow1 = &hsh[1][ln * NHID + (hi ? 8 : 0)];
  v16bf a0 = load_a_frag(arow0);           // h(0), kstep 0
  v16bf a1 = load_a_frag(arow0 + 32);      // h(0), kstep 1

  // Prime i2h pipeline for t=0.
  const float* ib0 = i2h + ((size_t)b0 * 256) + (size_t)hi * 8 * 256 + w * 16 + ln;
  v8f inx[4];
#pragma unroll
  for (int g = 0; g < 4; ++g)
#pragma unroll
    for (int j = 0; j < 8; ++j) inx[g][j] = ib0[(size_t)j * 256 + g * 64];

  v8f h1;
#pragma unroll
  for (int j = 0; j < 8; ++j) h1[j] = 0.0f;

  for (int t = 0; t < SLEN; ++t) {
    v8f icur[4];
#pragma unroll
    for (int g = 0; g < 4; ++g) icur[g] = inx[g];

    // Issue next step's i2h loads early (overlap with WMMA + barrier).
    int tn = (t + 1 < SLEN) ? t + 1 : t;
    const float* ibn = i2h + (size_t)(tn * BSZ + b0) * 256 +
                       (size_t)hi * 8 * 256 + w * 16 + ln;
#pragma unroll
    for (int g = 0; g < 4; ++g)
#pragma unroll
      for (int j = 0; j < 8; ++j) inx[g][j] = ibn[(size_t)j * 256 + g * 64];

    // gates = i2h + h @ W_glt  (accumulator initialized with i2h)
    v8f g4[4];
#pragma unroll
    for (int g = 0; g < 4; ++g) {
      v8f acc = icur[g];
      acc = wmma_bf16(a0, bglt[g][0], acc);
      acc = wmma_bf16(a1, bglt[g][1], acc);
      g4[g] = acc;
    }

#pragma unroll
    for (int j = 0; j < 8; ++j) {
      float fg = fast_sig(g4[0][j]);
      float gg = fast_tanh(g4[1][j]);
      float ig = fast_sig(g4[2][j]);
      float og = fast_sig(g4[3][j]);
      float cn = fg * c[j] + ig * gg;
      c[j]  = cn;
      h1[j] = og * fast_tanh(cn);
    }

    // h(t+1) into the other buffer: writers never touch the buffer laggard
    // readers may still be loading from -> a single barrier per step suffices.
    bf16* wbuf = &hsh[(t + 1) & 1][0];
#pragma unroll
    for (int j = 0; j < 8; ++j) {
      int row = hi ? 8 + j : j;
      wbuf[row * NHID + w * 16 + ln] = (bf16)h1[j];
    }
    __syncthreads();
    const bf16* ar = ((t + 1) & 1) ? arow1 : arow0;
    a0 = load_a_frag(ar);
    a1 = load_a_frag(ar + 32);

    if (w == 0) {                          // fused decoder: hs[t] @ W_dec + b_dec
      v8f d;
#pragma unroll
      for (int j = 0; j < 8; ++j) d[j] = decb;
      d = wmma_bf16(a0, bdec0, d);
      d = wmma_bf16(a1, bdec1, d);
      if (ln < 3) {
#pragma unroll
        for (int j = 0; j < 8; ++j) {
          int row = hi ? 8 + j : j;
          out[(size_t)(t * BSZ + b0 + row) * 3 + ln] = d[j];
        }
      }
    }
  }

  // Final hT, cT.
#pragma unroll
  for (int j = 0; j < 8; ++j) {
    int row = b0 + (hi ? 8 + j : j);
    out[OUT_HT + (size_t)row * NHID + w * 16 + ln] = h1[j];
    out[OUT_CT + (size_t)row * NHID + w * 16 + ln] = c[j];
  }
}

extern "C" void kernel_launch(void* const* d_in, const int* in_sizes, int n_in,
                              void* d_out, int out_size, void* d_ws, size_t ws_size,
                              hipStream_t stream) {
  const float* x     = (const float*)d_in[0];
  const float* h0    = (const float*)d_in[1];
  const float* c0    = (const float*)d_in[2];
  const float* W_pt0 = (const float*)d_in[3];
  const float* b_pt0 = (const float*)d_in[4];
  const float* W_pt1 = (const float*)d_in[5];
  const float* b_pt1 = (const float*)d_in[6];
  const float* W_glt = (const float*)d_in[7];
  const float* W_dec = (const float*)d_in[8];
  const float* b_dec = (const float*)d_in[9];
  float* out = (float*)d_out;
  char*  ws  = (char*)d_ws;

  pack_weights<<<dim3(389), dim3(256), 0, stream>>>(W_pt0, b_pt0, W_pt1, b_pt1,
                                                    W_glt, W_dec, ws);
  i2h_gemm<<<dim3(2048), dim3(128), 0, stream>>>(x, ws);
  lstm_scan<<<dim3(16), dim3(128), 0, stream>>>(h0, c0, b_dec, ws, out);
}
